// EnhancedTFNLayer_25331717112435
// MI455X (gfx1250) — matrix operand order repaired and checked by reading, not compiled, not run on hardware
//
#include <hip/hip_runtime.h>

// EnhancedTFNLayer for MI455X (gfx1250, wave32, WMMA f32<-f16 16x16x32)
// B=16, N=4096, G=1024, D=256, 4 diffusion steps.

#define B_  16
#define N_  4096
#define G_  1024
#define D_  256
#define DT_ 0.01f
#define EPS_ 1e-5f

typedef __attribute__((ext_vector_type(16))) _Float16 v16h;
typedef __attribute__((ext_vector_type(8)))  float    v8f;

// ---------------------------------------------------------------------------
// Kernel 1: FieldProjector.  field[b,g,d] = sum_n exp(-(p_n-grid_g)^2/(2s^2)) * emb[b,n,d]
// Per-block: 64 G-rows x 256 D-cols, K(=N) consumed 32 at a time via WMMA.
// ---------------------------------------------------------------------------
__global__ __launch_bounds__(256) void tfn_projector(
    const float* __restrict__ emb, const float* __restrict__ pos,
    const float* __restrict__ gridp, const float* __restrict__ sigma,
    float* __restrict__ field)
{
    __shared__ _Float16 ldsA[64 * 32];     // kern tile, row-major [g][k]
    __shared__ _Float16 ldsBt[256 * 32];   // emb tile, transposed [d][k]

    const int tid = threadIdx.x;
    const int b  = blockIdx.x >> 4;           // G_/64 = 16 tiles per batch
    const int g0 = (blockIdx.x & 15) * 64;

    const float sg = sigma[0];
    const float inv2s2 = 1.0f / (2.0f * sg * sg);

    const int lane = tid & 31, wave = tid >> 5;
    const int wi = wave >> 1, wj = wave & 1;   // 4x2 wave grid
    const int m16 = lane & 15, kh = lane >> 4;

    v8f c[8] = {};

    // A-staging assignment: each thread produces 8 kern values of one row
    const int arow = (tid * 8) >> 5;          // 0..63
    const int acol = (tid * 8) & 31;          // 0,8,16,24
    const float gval = gridp[(size_t)b * G_ + g0 + arow];

    for (int n0 = 0; n0 < N_; n0 += 32) {
        // ---- stage A: kern[g][k] = exp(-(p-g)^2 * inv2s2), computed on the fly
        #pragma unroll
        for (int i = 0; i < 8; ++i) {
            float pn = pos[(size_t)b * N_ + n0 + acol + i];
            float d  = pn - gval;
            ldsA[arow * 32 + acol + i] = (_Float16)__expf(-d * d * inv2s2);
        }
        // ---- stage B (transposed): ldsBt[d][k] = emb[b, n0+k, d], d = tid
        const float* esrc = &emb[((size_t)b * N_ + n0) * D_ + tid];
        #pragma unroll 4
        for (int k = 0; k < 32; ++k)
            ldsBt[tid * 32 + k] = (_Float16)esrc[(size_t)k * D_];
        if (n0 + 32 < N_)
            __builtin_prefetch(&emb[((size_t)b * N_ + n0 + 32) * D_ + tid], 0, 1);
        __syncthreads();

        // ---- A fragment (16x32 f16): lane<16 -> K 0..15 ; lane>=16 -> K 16..31
        v16h a;
        const _Float16* pa = &ldsA[(wi * 16 + m16) * 32 + kh * 16];
        #pragma unroll
        for (int i = 0; i < 8; ++i) { a[i] = pa[i]; a[i + 8] = pa[8 + i]; }

        // ---- 8 WMMAs covering this wave's 16x128 sub-tile
        #pragma unroll
        for (int j = 0; j < 8; ++j) {
            v16h bf = *(const v16h*)&ldsBt[(wj * 128 + j * 16 + m16) * 32 + kh * 16];
            c[j] = __builtin_amdgcn_wmma_f32_16x16x32_f16(
                       false, a, false, bf, (short)0, c[j], false, false);
        }
        __syncthreads();
    }

    // ---- store C (16x16 f32 layout: lane<16 M=r, lane>=16 M=8+r; N=lane&15)
    #pragma unroll
    for (int j = 0; j < 8; ++j) {
        const int e = wj * 128 + j * 16 + m16;
        #pragma unroll
        for (int r = 0; r < 8; ++r) {
            const int g = g0 + wi * 16 + kh * 8 + r;
            field[((size_t)b * G_ + g) * D_ + e] = c[j][r];
        }
    }
}

// ---------------------------------------------------------------------------
// Kernel 2: one diffusion step.
// fout = fin + DT*(alpha*lap(fin) + tanh(fin @ W_int + b_int))
// ---------------------------------------------------------------------------
__global__ __launch_bounds__(256) void tfn_dynamics_step(
    const float* __restrict__ fin, float* __restrict__ fout,
    const float* __restrict__ Wint, const float* __restrict__ bint,
    const float* __restrict__ alphap)
{
    __shared__ _Float16 ldsA[64 * 32];     // field tile [g][k]
    __shared__ _Float16 ldsBt[256 * 32];   // W_int tile transposed [e][k]

    const int tid = threadIdx.x;
    const int b  = blockIdx.x >> 4;
    const int g0 = (blockIdx.x & 15) * 64;
    const float alpha = alphap[0];

    const int lane = tid & 31, wave = tid >> 5;
    const int wi = wave >> 1, wj = wave & 1;
    const int m16 = lane & 15, kh = lane >> 4;

    v8f c[8] = {};

    const int am = tid >> 2;             // 0..63
    const int ak = (tid & 3) * 8;        // 0,8,16,24

    for (int kk = 0; kk < 8; ++kk) {     // K = D_ = 256 in chunks of 32
        const float* as = &fin[((size_t)b * G_ + g0 + am) * D_ + kk * 32 + ak];
        #pragma unroll
        for (int i = 0; i < 8; ++i)
            ldsA[am * 32 + ak + i] = (_Float16)as[i];
        const float* ws = &Wint[(size_t)(kk * 32) * D_ + tid];
        #pragma unroll 4
        for (int k = 0; k < 32; ++k)
            ldsBt[tid * 32 + k] = (_Float16)ws[(size_t)k * D_];
        __syncthreads();

        v16h a;
        const _Float16* pa = &ldsA[(wi * 16 + m16) * 32 + kh * 16];
        #pragma unroll
        for (int i = 0; i < 8; ++i) { a[i] = pa[i]; a[i + 8] = pa[8 + i]; }

        #pragma unroll
        for (int j = 0; j < 8; ++j) {
            v16h bf = *(const v16h*)&ldsBt[(wj * 128 + j * 16 + m16) * 32 + kh * 16];
            c[j] = __builtin_amdgcn_wmma_f32_16x16x32_f16(
                       false, a, false, bf, (short)0, c[j], false, false);
        }
        __syncthreads();
    }

    // ---- elementwise epilogue: Laplacian (edge-clamped) + tanh + Euler step
    const size_t rowb = (size_t)b * G_ * D_;
    #pragma unroll
    for (int j = 0; j < 8; ++j) {
        const int e = wj * 128 + j * 16 + m16;
        const float be = bint[e];
        #pragma unroll
        for (int r = 0; r < 8; ++r) {
            const int g = g0 + wi * 16 + kh * 8 + r;
            const size_t idx = rowb + (size_t)g * D_ + e;
            float f  = fin[idx];
            float fm = fin[idx - (g > 0       ? (size_t)D_ : 0)];
            float fp = fin[idx + (g < G_ - 1  ? (size_t)D_ : 0)];
            float lap   = fm - 2.0f * f + fp;
            float inter = tanhf(c[j][r] + be);
            fout[idx] = f + DT_ * (alpha * lap + inter);
        }
    }
}

// ---------------------------------------------------------------------------
// Kernel 3: sample + residual + LN1 + (enhanced @ W_out + b_out) + residual + LN2
// 64 tokens per block, all fused in LDS (enhanced never touches HBM).
// ---------------------------------------------------------------------------
__global__ __launch_bounds__(256) void tfn_sample_ln_out(
    const float* __restrict__ field, const float* __restrict__ emb,
    const float* __restrict__ pos,
    const float* __restrict__ Wout, const float* __restrict__ bout,
    const float* __restrict__ ln1g, const float* __restrict__ ln1b,
    const float* __restrict__ ln2g, const float* __restrict__ ln2b,
    float* __restrict__ out)
{
    extern __shared__ char smem[];
    float*    ldsE  = (float*)smem;               // 64 x 256 f32 (enhanced)
    float*    ldsY  = ldsE + 64 * D_;             // 64 x 256 f32 (pre-LN2)
    _Float16* ldsBt = (_Float16*)(ldsY + 64 * D_);// 256 x 32 f16 (W_out chunk, transposed)
    float*    mu    = (float*)(ldsBt + D_ * 32);  // 64
    float*    rs    = mu + 64;                    // 64

    const int tid = threadIdx.x;
    const int b  = blockIdx.x >> 6;               // N_/64 = 64 tiles per batch
    const int n0 = (blockIdx.x & 63) * 64;

    const int lane = tid & 31, wave = tid >> 5;
    const int wi = wave >> 1, wj = wave & 1;
    const int m16 = lane & 15, kh = lane >> 4;

    // ---- phase 1: sample field at token positions + residual
    const float g1 = ln1g[tid], b1 = ln1b[tid];
    for (int m = 0; m < 64; ++m) {
        const int n = n0 + m;
        float u  = pos[(size_t)b * N_ + n] * (float)(G_ - 1);
        int i0 = (int)floorf(u);
        i0 = i0 < 0 ? 0 : (i0 > G_ - 2 ? G_ - 2 : i0);
        float w = u - (float)i0;
        const size_t fb = ((size_t)b * G_ + i0) * D_ + tid;
        float f0 = field[fb], f1 = field[fb + D_];
        ldsE[m * D_ + tid] = (1.0f - w) * f0 + w * f1
                           + emb[((size_t)b * N_ + n) * D_ + tid];
    }
    __syncthreads();

    // ---- LN1 row statistics (one thread per row)
    if (tid < 64) {
        float s = 0.0f, q = 0.0f;
        for (int d = 0; d < D_; ++d) { float x = ldsE[tid * D_ + d]; s += x; q += x * x; }
        float mean = s * (1.0f / D_);
        float var  = q * (1.0f / D_) - mean * mean;
        mu[tid] = mean;
        rs[tid] = rsqrtf(var + EPS_);
    }
    __syncthreads();
    for (int m = 0; m < 64; ++m)
        ldsE[m * D_ + tid] = (ldsE[m * D_ + tid] - mu[m]) * rs[m] * g1 + b1;
    __syncthreads();

    // ---- phase 2: GEMM enhanced @ W_out via WMMA (A frags converted from f32 LDS)
    v8f c[8] = {};
    for (int kk = 0; kk < 8; ++kk) {
        __syncthreads();                       // previous chunk's B frags consumed
        const float* ws = &Wout[(size_t)(kk * 32) * D_ + tid];
        #pragma unroll 4
        for (int k = 0; k < 32; ++k)
            ldsBt[tid * 32 + k] = (_Float16)ws[(size_t)k * D_];
        __syncthreads();

        v16h a;
        const float* pe = &ldsE[(wi * 16 + m16) * D_ + kk * 32 + kh * 16];
        #pragma unroll
        for (int i = 0; i < 8; ++i) {
            a[i]     = (_Float16)pe[i];
            a[i + 8] = (_Float16)pe[8 + i];
        }
        #pragma unroll
        for (int j = 0; j < 8; ++j) {
            v16h bf = *(const v16h*)&ldsBt[(wj * 128 + j * 16 + m16) * 32 + kh * 16];
            c[j] = __builtin_amdgcn_wmma_f32_16x16x32_f16(
                       false, a, false, bf, (short)0, c[j], false, false);
        }
    }
    __syncthreads();

    // ---- phase 3: y = out + b_out + enhanced  -> LDS, then LN2 -> global
    #pragma unroll
    for (int j = 0; j < 8; ++j) {
        const int e = wj * 128 + j * 16 + m16;
        const float be = bout[e];
        #pragma unroll
        for (int r = 0; r < 8; ++r) {
            const int m = wi * 16 + kh * 8 + r;
            ldsY[m * D_ + e] = c[j][r] + be + ldsE[m * D_ + e];
        }
    }
    __syncthreads();
    if (tid < 64) {
        float s = 0.0f, q = 0.0f;
        for (int d = 0; d < D_; ++d) { float x = ldsY[tid * D_ + d]; s += x; q += x * x; }
        float mean = s * (1.0f / D_);
        float var  = q * (1.0f / D_) - mean * mean;
        mu[tid] = mean;
        rs[tid] = rsqrtf(var + EPS_);
    }
    __syncthreads();
    const float g2 = ln2g[tid], b2 = ln2b[tid];
    for (int m = 0; m < 64; ++m)
        out[((size_t)b * N_ + n0 + m) * D_ + tid] =
            (ldsY[m * D_ + tid] - mu[m]) * rs[m] * g2 + b2;
}

// ---------------------------------------------------------------------------
extern "C" void kernel_launch(void* const* d_in, const int* in_sizes, int n_in,
                              void* d_out, int out_size, void* d_ws, size_t ws_size,
                              hipStream_t stream)
{
    (void)in_sizes; (void)n_in; (void)out_size; (void)ws_size;
    const float* emb   = (const float*)d_in[0];
    const float* pos   = (const float*)d_in[1];
    const float* gridp = (const float*)d_in[2];
    const float* sigma = (const float*)d_in[3];
    const float* alpha = (const float*)d_in[4];
    const float* Wint  = (const float*)d_in[5];
    const float* bint  = (const float*)d_in[6];
    const float* Wout  = (const float*)d_in[7];
    const float* bout  = (const float*)d_in[8];
    const float* ln1g  = (const float*)d_in[9];
    const float* ln1b  = (const float*)d_in[10];
    const float* ln2g  = (const float*)d_in[11];
    const float* ln2b  = (const float*)d_in[12];
    float* outp = (float*)d_out;

    float* fA = (float*)d_ws;                       // 16 MB
    float* fB = fA + (size_t)B_ * G_ * D_;          // 16 MB (ping-pong)

    dim3 blk(256);
    tfn_projector<<<dim3(B_ * (G_ / 64)), blk, 0, stream>>>(emb, pos, gridp, sigma, fA);

    float* src = fA; float* dst = fB;
    for (int s = 0; s < 4; ++s) {
        tfn_dynamics_step<<<dim3(B_ * (G_ / 64)), blk, 0, stream>>>(src, dst, Wint, bint, alpha);
        float* t = src; src = dst; dst = t;         // final field ends up in src
    }

    const size_t smembytes = (size_t)(2 * 64 * D_) * sizeof(float)   // ldsE + ldsY
                           + (size_t)(D_ * 32) * sizeof(_Float16)    // ldsBt
                           + (size_t)(2 * 64) * sizeof(float);       // mu + rs
    tfn_sample_ln_out<<<dim3(B_ * (N_ / 64)), blk, smembytes, stream>>>(
        src, emb, pos, Wout, bout, ln1g, ln1b, ln2g, ln2b, outp);
}
